// TopNGating_21912923144949
// MI455X (gfx1250) — matrix operand order repaired and checked by reading, not compiled, not run on hardware
//
#include <hip/hip_runtime.h>

#define B_    4
#define N_    4096
#define DIM_  1024
#define E_    8
#define CAP_  640
#define BN_   (B_ * N_)
#define BNEC_ ((size_t)BN_ * E_ * CAP_)
#define EPS_  1e-9f

typedef __attribute__((ext_vector_type(16))) _Float16 v16h;
typedef __attribute__((ext_vector_type(8)))  float    v8f;
typedef __attribute__((ext_vector_type(4)))  float    v4f;

// workspace layout (float offsets)
#define WS_DENS 0                  // [B*E] top-1 expert counts
#define WS_PROX 32                 // [B*E] sum of softmax probs
#define WS_Z    64                 // [1]   sum of lse^2
#define WS_E0   96                 // [BN] int
#define WS_E1   (WS_E0 + BN_)      // [BN] int
#define WS_G0   (WS_E1 + BN_)      // [BN] float
#define WS_G1   (WS_G0 + BN_)      // [BN] float
#define WS_M1   (WS_G1 + BN_)      // [BN] int (routed flag k=1)
#define WS_P0   (WS_M1 + BN_)      // [BN] int
#define WS_K0   (WS_P0 + BN_)      // [BN] int
#define WS_P1   (WS_K0 + BN_)      // [BN] int
#define WS_K1   (WS_P1 + BN_)      // [BN] int

__global__ void init_acc_kernel(float* __restrict__ ws) {
    int t = threadIdx.x;
    if (t < 65) ws[t] = 0.0f;
}

// one wave per 16-token tile: WMMA logits -> softmax -> top2 -> routing flags
__global__ __launch_bounds__(32) void gate_kernel(const float* __restrict__ x,
                                                  const float* __restrict__ w,
                                                  const float* __restrict__ probs,
                                                  float* __restrict__ ws) {
    __shared__ float lt[256];
    __shared__ float sdens[E_];
    __shared__ float sprox[E_];
    __shared__ float sz;

    const int lane = threadIdx.x;
    if (lane < E_) { sdens[lane] = 0.0f; sprox[lane] = 0.0f; }
    if (lane == 0) sz = 0.0f;

    const int tile = blockIdx.x;        // BN_/16 = 1024 tiles
    const int b    = tile >> 8;         // 256 tiles per batch
    const int n0   = (tile & 255) << 4;

    const int m     = lane & 15;
    const int hi    = lane >> 4;        // 0 or 1
    const int abase = hi ? 16 : 0;      // A-fragment K base (contiguous, matches B)
    const int kbase = hi ? 16 : 0;      // B-fragment K base (contiguous, matches A)
    const int ncol  = lane & 15;        // B/C column

    const float* xrow = x + (size_t)(b * N_ + n0 + m) * DIM_;

    v8f c = {};
    for (int kk = 0; kk < DIM_ / 32; ++kk) {
        const float* xp = xrow + kk * 32;
        v16h a, bf;
#pragma unroll
        for (int i = 0; i < 8; ++i) {
            a[i]     = (_Float16)xp[abase + i];          // K = abase..abase+7
            a[8 + i] = (_Float16)xp[abase + 8 + i];      // K = abase+8..abase+15
        }
        if (ncol < E_) {
            const float* wp = w + (size_t)ncol * DIM_ + kk * 32 + kbase;
#pragma unroll
            for (int i = 0; i < 16; ++i) bf[i] = (_Float16)wp[i];   // K = kbase+i
        } else {
#pragma unroll
            for (int i = 0; i < 16; ++i) bf[i] = (_Float16)0.0f;    // padded experts
        }
        c = __builtin_amdgcn_wmma_f32_16x16x32_f16(false, a, false, bf,
                                                   (short)0, c, false, false);
    }
    // C layout: lane l holds column l&15, rows v + 8*(l>=16)
#pragma unroll
    for (int v = 0; v < 8; ++v) lt[(v + hi * 8) * 16 + ncol] = c[v];
    __syncthreads();

    if (lane < 16) {
        float l[E_], p[E_];
#pragma unroll
        for (int e = 0; e < E_; ++e) l[e] = lt[lane * 16 + e];
        float mx = l[0];
#pragma unroll
        for (int e = 1; e < E_; ++e) mx = fmaxf(mx, l[e]);
        float se = 0.0f;
#pragma unroll
        for (int e = 0; e < E_; ++e) { p[e] = expf(l[e] - mx); se += p[e]; }
        const float lse = logf(se) + mx;                 // logsumexp for z-loss
        const float inv = 1.0f / se;
#pragma unroll
        for (int e = 0; e < E_; ++e) p[e] *= inv;        // raw_gates (softmax)

        // top-1 (strict > keeps lowest index, matching jax top_k tie order)
        int e0 = 0; float v0 = p[0];
#pragma unroll
        for (int e = 1; e < E_; ++e) if (p[e] > v0) { v0 = p[e]; e0 = e; }
        // top-2
        int e1i = (e0 == 0) ? 1 : 0; float v1 = p[e1i];
#pragma unroll
        for (int e = 0; e < E_; ++e)
            if (e != e0 && p[e] > v1) { v1 = p[e]; e1i = e; }

        float denom = v0 + v1;
        if (denom < EPS_) denom = EPS_;
        const float g0 = v0 / denom;
        const float g1 = v1 / denom;

        const int gidx = b * N_ + n0 + lane;
        // should_route[1] = probs[1,b,n] < g1/0.2 ; should_route[0] forced true
        const float pr1 = probs[BN_ + gidx];
        const int m1 = (pr1 < g1 * 5.0f) ? 1 : 0;

        ((int*)(ws + WS_E0))[gidx]   = e0;
        ((int*)(ws + WS_E1))[gidx]   = e1i;
        (ws + WS_G0)[gidx]           = g0;
        (ws + WS_G1)[gidx]           = g1;
        ((int*)(ws + WS_M1))[gidx]   = m1;

        atomicAdd(&sdens[e0], 1.0f);                     // mask_1 (pre-route) density
#pragma unroll
        for (int e = 0; e < E_; ++e) atomicAdd(&sprox[e], p[e]);
        atomicAdd(&sz, lse * lse);
    }
    __syncthreads();
    if (lane < E_) {
        atomicAdd(&ws[WS_DENS + b * E_ + lane], sdens[lane]);
        atomicAdd(&ws[WS_PROX + b * E_ + lane], sprox[lane]);
    }
    if (lane == 0) atomicAdd(&ws[WS_Z], sz);
}

// one wave32 per (b,e): ballot-prefix scan -> capacity positions, k=0 then k=1
__global__ __launch_bounds__(32) void scan_kernel(float* __restrict__ ws) {
    const int b = blockIdx.x >> 3;
    const int e = blockIdx.x & 7;
    const int lane = threadIdx.x;

    const int* e0p = (const int*)(ws + WS_E0);
    const int* e1p = (const int*)(ws + WS_E1);
    const int* m1p = (const int*)(ws + WS_M1);
    int* p0p = (int*)(ws + WS_P0);
    int* k0p = (int*)(ws + WS_K0);
    int* p1p = (int*)(ws + WS_P1);
    int* k1p = (int*)(ws + WS_K1);

    const unsigned lmask = (lane == 0) ? 0u : (0xffffffffu >> (32 - lane));

    // k = 0: always routed; rank among tokens assigned to expert e
    unsigned run = 0;
    for (int c0 = 0; c0 < N_; c0 += 32) {
        const int gi = b * N_ + c0 + lane;
        const bool f = (e0p[gi] == e);
        const unsigned msk = __builtin_amdgcn_ballot_w32(f);
        const unsigned pos = run + (unsigned)__builtin_popcount(msk & lmask);
        if (f) { p0p[gi] = (int)pos; k0p[gi] = (pos < CAP_) ? 1 : 0; }
        run += (unsigned)__builtin_popcount(msk);
    }
    const unsigned base1 = (run < CAP_) ? run : CAP_;   // prev_expert_count (capped)

    // k = 1: routed iff m1; positions offset by capped k=0 count
    unsigned run1 = 0;
    for (int c0 = 0; c0 < N_; c0 += 32) {
        const int gi = b * N_ + c0 + lane;
        const bool own = (e1p[gi] == e);
        const bool f = own && (m1p[gi] != 0);
        const unsigned msk = __builtin_amdgcn_ballot_w32(f);
        const unsigned pos = base1 + run1 + (unsigned)__builtin_popcount(msk & lmask);
        if (own) { p1p[gi] = (int)pos; k1p[gi] = (f && pos < CAP_) ? 1 : 0; }
        run1 += (unsigned)__builtin_popcount(msk);
    }
}

// bulk zero of dispatch+combine (671 MB) with nontemporal 128-bit stores
__global__ void zerofill_kernel(float* __restrict__ out, size_t n4) {
    size_t i = (size_t)blockIdx.x * blockDim.x + threadIdx.x;
    const size_t stride = (size_t)gridDim.x * blockDim.x;
    const v4f z = {0.f, 0.f, 0.f, 0.f};
    v4f* o4 = (v4f*)out;
    for (; i < n4; i += stride) __builtin_nontemporal_store(z, &o4[i]);
}

// scatter <=2 nonzeros per token + write loss scalars
__global__ void scatter_kernel(const float* __restrict__ ws, float* __restrict__ out) {
    const int tid = blockIdx.x * blockDim.x + threadIdx.x;
    if (tid < 2 * BN_) {
        const int k = tid >> 14;           // BN_ = 16384
        const int r = tid & (BN_ - 1);
        const int keep = k ? ((const int*)(ws + WS_K1))[r] : ((const int*)(ws + WS_K0))[r];
        if (keep) {
            const int e   = k ? ((const int*)(ws + WS_E1))[r] : ((const int*)(ws + WS_E0))[r];
            const int pos = k ? ((const int*)(ws + WS_P1))[r] : ((const int*)(ws + WS_P0))[r];
            const float g = k ? (ws + WS_G1)[r] : (ws + WS_G0)[r];
            const size_t idx = ((size_t)r * E_ + e) * CAP_ + pos;
            out[idx] = 1.0f;               // dispatch
            out[BNEC_ + idx] = g;          // combine
        }
    }
    if (tid == 0) {
        float s = 0.0f;
        for (int i = 0; i < B_ * E_; ++i) s += ws[WS_PROX + i] * ws[WS_DENS + i];
        // mean_be(proxy/N * dens/N) * E^2 = s * E^2 / (B*E*N*N) = s*2/N^2
        out[2 * BNEC_]     = s * 2.0f / ((float)N_ * (float)N_);
        out[2 * BNEC_ + 1] = ws[WS_Z] / (float)BN_;
    }
}

extern "C" void kernel_launch(void* const* d_in, const int* in_sizes, int n_in,
                              void* d_out, int out_size, void* d_ws, size_t ws_size,
                              hipStream_t stream) {
    const float* x     = (const float*)d_in[0];   // (B,N,DIM) f32
    const float* w     = (const float*)d_in[1];   // (E,DIM)   f32
    const float* probs = (const float*)d_in[2];   // (TOPN,B,N) f32
    float* out = (float*)d_out;
    float* ws  = (float*)d_ws;

    (void)in_sizes; (void)n_in; (void)out_size; (void)ws_size;

    // 1) zero loss accumulators (ws is not re-poisoned between replays)
    init_acc_kernel<<<1, 128, 0, stream>>>(ws);

    // 2) bulk zero-fill of dispatch+combine (2*BNEC floats, divisible by 4)
    const size_t n4 = (2 * BNEC_) / 4;
    zerofill_kernel<<<8192, 256, 0, stream>>>(out, n4);

    // 3) WMMA gating: 1024 tiles of 16 tokens, one wave each
    gate_kernel<<<BN_ / 16, 32, 0, stream>>>(x, w, probs, ws);

    // 4) capacity scan: one wave per (b,e)
    scan_kernel<<<B_ * E_, 32, 0, stream>>>(ws);

    // 5) scatter nonzeros + loss scalars
    scatter_kernel<<<(2 * BN_ + 255) / 256, 256, 0, stream>>>(ws, out);
}
